// CSSRClassifier_87024627352150
// MI455X (gfx1250) — matrix operand order repaired and checked by reading, not compiled, not run on hardware
//
#include <hip/hip_runtime.h>

// ---------------------------------------------------------------------------
// CSSR classifier fused kernel for MI455X (gfx1250, wave32, WMMA bf16).
//
// ~37 GFLOP of GEMM vs ~30MB unique bytes -> matrix-core bound. bf16
// operands / f32 accumulate via v_wmma_f32_16x16x32_bf16; the whole
// 3-GEMM + tanh + prototype-distance chain is fused in LDS. tanh uses the
// CDNA5 V_TANH_F32 transcendental. GEMM1's K-loop is software-pipelined with
// double-buffered LDS chunks fed by GLOBAL_LOAD_ASYNC_TO_LDS_B128 (ASYNCcnt).
// ---------------------------------------------------------------------------

typedef __attribute__((ext_vector_type(16))) __bf16 v16bf;
typedef __attribute__((ext_vector_type(8)))  __bf16 v8bf;
typedef __attribute__((ext_vector_type(8)))  float  v8f;
typedef __attribute__((ext_vector_type(4)))  float  v4f;
typedef int v4i_vs __attribute__((vector_size(16)));   // matches builtin proto

// Problem constants (from reference)
constexpr int Bn   = 64;           // batch
constexpr int Cn   = 512;          // in channels
constexpr int Pn   = 64;           // H*W
constexpr int Nn   = Bn * Pn;      // 4096 GEMM columns
constexpr int KCLS = 100;          // classes
constexpr int H1   = 64;
constexpr int HID  = 128;
constexpr int Ln   = 32;
constexpr int NT   = 128;          // N-tile per block (covers exactly 2 b's)
constexpr float REDC = -0.1f;
constexpr float CLIP = 100.0f;
constexpr float PUSH_THRESH = 10000.0f;

constexpr int LOGITS1_OFF = Bn * KCLS * Pn;          // 409600
constexpr int SCALAR_OFF  = 2 * Bn * KCLS * Pn;      // 819200

// LDS layout (bytes), all segments 16B aligned. sX/sW1 are double buffered.
constexpr int XBUF = NT * 64;               // elements per X chunk buffer
constexpr int WBUF = H1 * 64;               // elements per W1 chunk buffer
constexpr int OFF_SX  = 0;                      // 2*[NT][64] bf16
constexpr int OFF_SW1 = OFF_SX  + 2*XBUF*2;     // 2*[64][64] bf16
constexpr int OFF_SH  = OFF_SW1 + 2*WBUF*2;     // [NT][H1]   bf16
constexpr int OFF_SW2 = OFF_SH  + NT*H1*2;      // [HID][H1]  bf16
constexpr int OFF_SH2 = OFF_SW2 + HID*H1*2;     // [NT][HID]  bf16
constexpr int OFF_SW3 = OFF_SH2 + NT*HID*2;     // [Ln][HID]  bf16
constexpr int OFF_SP  = OFF_SW3 + Ln*HID*2;     // [Ln][Pn]   f32
constexpr int OFF_SP1 = OFF_SP  + Ln*Pn*4;      // [Ln][Pn]   f32
constexpr int OFF_RED = OFF_SP1 + Ln*Pn*4;      // 4 floats
constexpr int SMEM_BYTES = OFF_RED + 64;        // ~139.3 KB < 320 KB WGP LDS

// Hardware tanh (CDNA5 V_TANH_F32); guarded fallbacks.
__device__ __forceinline__ float fast_tanhf(float x) {
#if __has_builtin(__builtin_amdgcn_tanhf)
    return __builtin_amdgcn_tanhf(x);
#elif __has_builtin(__builtin_amdgcn_tanh_f32)
    return __builtin_amdgcn_tanh_f32(x);
#elif defined(__AMDGCN__)
    float r;
    asm("v_tanh_f32 %0, %1" : "=v"(r) : "v"(x));
    return r;
#else
    return tanhf(x);
#endif
}

__device__ __forceinline__ void sched_fence() {
#if __has_builtin(__builtin_amdgcn_sched_barrier)
    __builtin_amdgcn_sched_barrier(0);   // keep DS-load group ahead of WMMAs
#endif
}

// ---- CDNA5 async global->LDS copy (GLOBAL_LOAD_ASYNC_TO_LDS_B128) ---------
#if __has_builtin(__builtin_amdgcn_global_load_async_to_lds_b128)
#define ASYNC_LDS_COPY 1
#endif

__device__ __forceinline__ void async_copy16(const void* g, void* l) {
#ifdef ASYNC_LDS_COPY
    typedef __attribute__((address_space(1))) v4i_vs* gp_t;   // global int4*
    typedef __attribute__((address_space(3))) v4i_vs* lp_t;   // LDS int4*
    __builtin_amdgcn_global_load_async_to_lds_b128(
        (gp_t)(unsigned long long)g,
        (lp_t)(unsigned int)(unsigned long long)l,
        0, 0);
#else
    *(v8bf*)l = *(const v8bf*)g;
#endif
}

__device__ __forceinline__ void async_fence() {
#ifdef ASYNC_LDS_COPY
#if __has_builtin(__builtin_amdgcn_s_wait_asynccnt)
    __builtin_amdgcn_s_wait_asynccnt(0);
#else
    asm volatile("s_wait_asynccnt 0x0" ::: "memory");
#endif
#endif
}

union FragU { v16bf v; v8bf h[2]; };

// A fragment (16x32 bf16, ISA 7.12.2 layout): lane m=lane&15, hi=lane>>4.
__device__ __forceinline__ v16bf load_frag_A(const __bf16* base, int ldk,
                                             int row, int koff, int hi) {
    const __bf16* p = base + row * ldk + koff + hi * 8;
    FragU u;
    u.h[0] = *(const v8bf*)(p);
    u.h[1] = *(const v8bf*)(p + 16);
    return u.v;
}

// B fragment (32x16 bf16): lane n=lane&15, hi=lane>>4; K-contiguous columns.
__device__ __forceinline__ v16bf load_frag_B(const __bf16* base, int ldk,
                                             int col, int koff, int hi) {
    const __bf16* p = base + col * ldk + koff + hi * 8;
    FragU u;
    u.h[0] = *(const v8bf*)(p);
    u.h[1] = *(const v8bf*)(p + 16);
    return u.v;
}

// --------------------------- pre-pass kernels ------------------------------

// x[B,C,P] f32 -> Xb[N][C] bf16 (K-contiguous rows; LDS-tiled 64x64 transpose)
__global__ void cssr_pack_x(const float* __restrict__ x, __bf16* __restrict__ xb) {
    __shared__ __bf16 tile[64 * 66];     // pad 66: conflict-free strided reads
    const int c0 = blockIdx.x * 64;      // C/64 = 8 tiles
    const int n0 = blockIdx.y * 64;      // N/64 = 64 tiles; b = blockIdx.y
    const int b  = n0 >> 6;
    #pragma unroll
    for (int it = 0; it < 16; ++it) {
        const int idx = threadIdx.x + it * 256;
        const int c = idx >> 6, p = idx & 63;         // read coalesced over p
        tile[c * 66 + p] = (__bf16)x[((b * Cn) + c0 + c) * Pn + p];
    }
    __syncthreads();
    #pragma unroll
    for (int it = 0; it < 16; ++it) {
        const int idx = threadIdx.x + it * 256;
        const int p = idx >> 6, c = idx & 63;         // write coalesced over c
        xb[(size_t)(n0 + p) * Cn + c0 + c] = tile[c * 66 + p];
    }
}

__global__ void cssr_cvt_bf16(const float* __restrict__ src,
                              __bf16* __restrict__ dst, int n) {
    for (int i = blockIdx.x * blockDim.x + threadIdx.x; i < n;
         i += gridDim.x * blockDim.x) {
        dst[i] = (__bf16)src[i];
    }
}

// Stage one 64-wide K-chunk of X[NT][64] and W1[64][64] into LDS buffers.
__device__ __forceinline__ void stage_chunk(const __bf16* __restrict__ Xb,
                                            const __bf16* __restrict__ W1b,
                                            __bf16* sXbuf, __bf16* sW1buf,
                                            int n0, int krow, int kc, int tid) {
#ifdef ASYNC_LDS_COPY
    #pragma unroll
    for (int it = 0; it < 4; ++it) {
        const int idx = tid + it * 256;
        const int n = idx >> 3, c = (idx & 7) * 8;
        async_copy16(&Xb[(size_t)(n0 + n) * Cn + kc + c], &sXbuf[n * 64 + c]);
    }
    #pragma unroll
    for (int it = 0; it < 2; ++it) {
        const int idx = tid + it * 256;
        const int o = idx >> 3, c = (idx & 7) * 8;
        async_copy16(&W1b[(size_t)(krow + o) * Cn + kc + c], &sW1buf[o * 64 + c]);
    }
#else
    v8bf tx[4], tw[2];
    #pragma unroll
    for (int it = 0; it < 4; ++it) {
        const int idx = tid + it * 256;
        const int n = idx >> 3, c = (idx & 7) * 8;
        tx[it] = *(const v8bf*)&Xb[(size_t)(n0 + n) * Cn + kc + c];
    }
    #pragma unroll
    for (int it = 0; it < 2; ++it) {
        const int idx = tid + it * 256;
        const int o = idx >> 3, c = (idx & 7) * 8;
        tw[it] = *(const v8bf*)&W1b[(size_t)(krow + o) * Cn + kc + c];
    }
    #pragma unroll
    for (int it = 0; it < 4; ++it) {
        const int idx = tid + it * 256;
        const int n = idx >> 3, c = (idx & 7) * 8;
        *(v8bf*)&sXbuf[n * 64 + c] = tx[it];
    }
    #pragma unroll
    for (int it = 0; it < 2; ++it) {
        const int idx = tid + it * 256;
        const int o = idx >> 3, c = (idx & 7) * 8;
        *(v8bf*)&sW1buf[o * 64 + c] = tw[it];
    }
#endif
}

// ------------------------------ main kernel --------------------------------
// grid = (N/NT = 32, K = 100), block = 256 threads (8 wave32)
__global__ __launch_bounds__(256)
void cssr_fused(const __bf16* __restrict__ Xb,     // [N][C]
                const __bf16* __restrict__ W1b,    // [K][H1][C]
                const __bf16* __restrict__ W2b,    // [K][HID][H1]
                const __bf16* __restrict__ W3b,    // [K][L][HID]
                const float*  __restrict__ proto,
                const float*  __restrict__ proto1,
                float* __restrict__ f,
                float* __restrict__ f1,
                float* __restrict__ out) {
    extern __shared__ char smem[];
    __bf16* sX  = (__bf16*)(smem + OFF_SX);
    __bf16* sW1 = (__bf16*)(smem + OFF_SW1);
    __bf16* sH  = (__bf16*)(smem + OFF_SH);
    __bf16* sW2 = (__bf16*)(smem + OFF_SW2);
    __bf16* sH2 = (__bf16*)(smem + OFF_SH2);
    __bf16* sW3 = (__bf16*)(smem + OFF_SW3);
    float*  sP   = (float*)(smem + OFF_SP);
    float*  sP1  = (float*)(smem + OFF_SP1);
    float*  sRed = (float*)(smem + OFF_RED);

    const int k    = blockIdx.y;
    const int krow = k * H1;
    const int n0   = blockIdx.x * NT;
    const int tid  = threadIdx.x;
    const int wave = tid >> 5;
    const int lane = tid & 31;
    const int m16  = lane & 15;
    const int hi   = lane >> 4;
    const int ncol = wave * 16 + m16;   // this lane's column within the N tile

    // Preload per-class weights & prototypes (async b128 copies when available)
    #pragma unroll
    for (int it = 0; it < 4; ++it) {
        const int i8 = (tid + it * 256) * 8;
        async_copy16(&W2b[(size_t)k * HID * H1 + i8], &sW2[i8]);
    }
    #pragma unroll
    for (int it = 0; it < 2; ++it) {
        const int i8 = (tid + it * 256) * 8;
        async_copy16(&W3b[(size_t)k * Ln * HID + i8], &sW3[i8]);
    }
    #pragma unroll
    for (int it = 0; it < 2; ++it) {
        const int i4 = (tid + it * 256) * 4;
        async_copy16(&proto [(size_t)k * Ln * Pn + i4], &sP [i4]);
        async_copy16(&proto1[(size_t)k * Ln * Pn + i4], &sP1[i4]);
    }
    if (tid < 4) sRed[tid] = 0.0f;

    // ---------------- GEMM1: h = tanh(W1[64,512] * X[512,NT]) ----------------
    // Software pipeline: stage chunk kc+64 (async) while computing chunk kc.
    v8f acc1[4];
    #pragma unroll
    for (int mt = 0; mt < 4; ++mt) acc1[mt] = (v8f){};

    stage_chunk(Xb, W1b, sX, sW1, n0, krow, 0, tid);
    async_fence();
    __syncthreads();

    for (int kc = 0; kc < Cn; kc += 64) {
        const int bufi = (kc >> 6) & 1;
        const __bf16* curX = sX  + bufi * XBUF;
        const __bf16* curW = sW1 + bufi * WBUF;
        if (kc + 64 < Cn)
            stage_chunk(Xb, W1b, sX + (bufi ^ 1) * XBUF, sW1 + (bufi ^ 1) * WBUF,
                        n0, krow, kc + 64, tid);
        #pragma unroll
        for (int ks = 0; ks < 2; ++ks) {
            const int koff = ks * 32;
            v16bf bfrag = load_frag_B(curX, 64, ncol, koff, hi);
            v16bf af[4];
            #pragma unroll
            for (int mt = 0; mt < 4; ++mt)
                af[mt] = load_frag_A(curW, 64, mt * 16 + m16, koff, hi);
            sched_fence();
            #pragma unroll
            for (int mt = 0; mt < 4; ++mt)
                acc1[mt] = __builtin_amdgcn_wmma_f32_16x16x32_bf16(
                    false, af[mt], false, bfrag, (short)0, acc1[mt], false, false);
        }
        async_fence();         // our next-chunk async copies have landed
        __syncthreads();       // everyone done reading cur / writing next
    }
    // D layout: lane holds column ncol, rows mt*16 + hi*8 + r.
    #pragma unroll
    for (int mt = 0; mt < 4; ++mt) {
        v8bf pk;
        #pragma unroll
        for (int r = 0; r < 8; ++r) pk[r] = (__bf16)fast_tanhf(acc1[mt][r]);
        *(v8bf*)&sH[ncol * H1 + mt * 16 + hi * 8] = pk;
    }
    __syncthreads();

    // ---------------- GEMM2: h2 = tanh(W2[128,64] * h[64,NT]) ----------------
    v8f acc2[8];
    #pragma unroll
    for (int mt = 0; mt < 8; ++mt) acc2[mt] = (v8f){};
    #pragma unroll
    for (int ks = 0; ks < 2; ++ks) {
        const int koff = ks * 32;
        v16bf bfrag = load_frag_B(sH, H1, ncol, koff, hi);
        #pragma unroll
        for (int g = 0; g < 2; ++g) {       // groups of 4 to cap VGPR pressure
            v16bf af[4];
            #pragma unroll
            for (int j = 0; j < 4; ++j)
                af[j] = load_frag_A(sW2, H1, (g * 4 + j) * 16 + m16, koff, hi);
            sched_fence();
            #pragma unroll
            for (int j = 0; j < 4; ++j)
                acc2[g * 4 + j] = __builtin_amdgcn_wmma_f32_16x16x32_bf16(
                    false, af[j], false, bfrag, (short)0, acc2[g * 4 + j],
                    false, false);
        }
    }
    #pragma unroll
    for (int mt = 0; mt < 8; ++mt) {
        v8bf pk;
        #pragma unroll
        for (int r = 0; r < 8; ++r) pk[r] = (__bf16)fast_tanhf(acc2[mt][r]);
        *(v8bf*)&sH2[ncol * HID + mt * 16 + hi * 8] = pk;
    }
    __syncthreads();

    // ---------------- GEMM3: lt = tanh(W3[32,128] * h2[128,NT]) --------------
    v8f acc3[2];
    acc3[0] = (v8f){}; acc3[1] = (v8f){};
    #pragma unroll
    for (int ks = 0; ks < 4; ++ks) {
        const int koff = ks * 32;
        v16bf bfrag = load_frag_B(sH2, HID, ncol, koff, hi);
        v16bf af[2];
        #pragma unroll
        for (int t = 0; t < 2; ++t)
            af[t] = load_frag_A(sW3, HID, t * 16 + m16, koff, hi);
        sched_fence();
        #pragma unroll
        for (int t = 0; t < 2; ++t)
            acc3[t] = __builtin_amdgcn_wmma_f32_16x16x32_bf16(
                false, af[t], false, bfrag, (short)0, acc3[t], false, false);
    }

    // ---------------- distances + clip + loss partials -----------------------
    const int p_idx = ncol & 63;
    const int b_loc = ncol >> 6;
    float s = 0.0f, s1 = 0.0f;
    #pragma unroll
    for (int t = 0; t < 2; ++t)
        #pragma unroll
        for (int r = 0; r < 8; ++r) {
            const int l = t * 16 + hi * 8 + r;
            const float lt = fast_tanhf(acc3[t][r]);
            const float d0 = lt - sP[l * Pn + p_idx];
            const float d1 = lt - sP1[l * Pn + p_idx];
            s  += d0 * d0;
            s1 += d1 * d1;
        }
    // lane j and j+16 hold complementary L-halves of the same column
    s  += __shfl_xor(s, 16, 32);
    s1 += __shfl_xor(s1, 16, 32);
    const float er  = fminf(fmaxf(s  * REDC, -CLIP), CLIP);
    const float er1 = fminf(fmaxf(s1 * REDC, -CLIP), CLIP);

    if (hi == 0) {
        const int ng = n0 + ncol;
        const int b  = ng >> 6;
        const int p  = ng & 63;
        out[(b * KCLS + k) * Pn + p]               = er;   // logits  [B,K,H,W]
        out[LOGITS1_OFF + (b * KCLS + k) * Pn + p] = er1;  // logits1 [B,K,H,W]
        atomicAdd(&sRed[b_loc],     er);                   // ds_add_f32
        atomicAdd(&sRed[2 + b_loc], er1);
    }
    __syncthreads();
    if (tid < 2) {   // this block covers b = blockIdx.x*2 + {0,1} completely
        f [k * Bn + blockIdx.x * 2 + tid] = sRed[tid];
        f1[k * Bn + blockIdx.x * 2 + tid] = sRed[2 + tid];
    }
}

// -------------------------- pull/push loss kernel --------------------------
__global__ void cssr_loss(const float* __restrict__ f, const float* __restrict__ f1,
                          const int* __restrict__ ycls, float* __restrict__ outpp) {
    __shared__ float rp[128];
    __shared__ float rq[128];
    const int t = threadIdx.x;
    float pk = 0.0f, qk = 0.0f;
    if (t < KCLS) {
        float s1 = 0.0f, sf = 0.0f;
        int neq = 0, ncomb = 0;
        for (int b = 0; b < Bn; ++b) {
            const float fe  = f [t * Bn + b];
            const float fe1 = f1[t * Bn + b];
            if (ycls[b] == t) { ++neq; s1 += fe1; }
            else if (fe < PUSH_THRESH) { ++ncomb; sf += fe; }
        }
        pk = neq   ? s1 / (float)neq   : 0.0f;
        qk = ncomb ? sf / (float)ncomb : 0.0f;
    }
    rp[t] = pk; rq[t] = qk;
    __syncthreads();
    for (int stride = 64; stride > 0; stride >>= 1) {
        if (t < stride) { rp[t] += rp[t + stride]; rq[t] += rq[t + stride]; }
        __syncthreads();
    }
    if (t == 0) { outpp[0] = rp[0]; outpp[1] = rq[0]; }
}

// ------------------------------- launcher ----------------------------------
extern "C" void kernel_launch(void* const* d_in, const int* in_sizes, int n_in,
                              void* d_out, int out_size, void* d_ws, size_t ws_size,
                              hipStream_t stream) {
    const float* x    = (const float*)d_in[0];
    const int*   ycls = (const int*)d_in[1];
    // d_in[2], d_in[3] (safe_margin/safe_distance) unused by the reference
    const float* W1 = (const float*)d_in[4];
    const float* W2 = (const float*)d_in[5];
    const float* W3 = (const float*)d_in[6];
    const float* pr  = (const float*)d_in[7];
    const float* pr1 = (const float*)d_in[8];
    float* out = (float*)d_out;

    // Workspace carve-up (~13.3 MB), 16B-aligned segments
    char* ws = (char*)d_ws;
    size_t o = 0;
    __bf16* Xb  = (__bf16*)(ws + o); o += (size_t)Nn * Cn * 2;          // 4,194,304
    __bf16* W1b = (__bf16*)(ws + o); o += (size_t)KCLS * H1 * Cn * 2;   // 6,553,600
    __bf16* W2b = (__bf16*)(ws + o); o += (size_t)KCLS * HID * H1 * 2;  // 1,638,400
    __bf16* W3b = (__bf16*)(ws + o); o += (size_t)KCLS * Ln * HID * 2;  //   819,200
    float*  f   = (float*) (ws + o); o += (size_t)KCLS * Bn * 4;
    float*  f1  = (float*) (ws + o); o += (size_t)KCLS * Bn * 4;

    cssr_pack_x<<<dim3(Cn / 64, Nn / 64), 256, 0, stream>>>(x, Xb);
    cssr_cvt_bf16<<<2048, 256, 0, stream>>>(W1, W1b, KCLS * H1 * Cn);
    cssr_cvt_bf16<<<1024, 256, 0, stream>>>(W2, W2b, KCLS * HID * H1);
    cssr_cvt_bf16<<<512, 256, 0, stream>>>(W3, W3b, KCLS * Ln * HID);

    dim3 grid(Nn / NT, KCLS);   // (32, 100)
    cssr_fused<<<grid, 256, SMEM_BYTES, stream>>>(Xb, W1b, W2b, W3b, pr, pr1,
                                                  f, f1, out);

    cssr_loss<<<1, 128, 0, stream>>>(f, f1, ycls, out + SCALAR_OFF);
}
